// Int8Conv1d_37580963840180
// MI455X (gfx1250) — matrix operand order repaired and checked by reading, compiled, not run on hardware
//
#include <hip/hip_runtime.h>

typedef _Float16 half_t;
typedef __attribute__((ext_vector_type(16))) _Float16 v16h;
typedef __attribute__((ext_vector_type(8)))  _Float16 v8h;
typedef __attribute__((ext_vector_type(8)))  float    v8f;

constexpr int BATCH = 16, CIN = 512, COUT = 512, KW = 3, LEN = 8192;
constexpr int TILE_CO = 128, TILE_L = 128, TILE_CI = 32;
constexpr int XROWS = TILE_L + 2;       // 130 l-positions (halo of 1 on each side)
constexpr int ROWP  = 40;               // padded LDS row stride in halves (80 B)
constexpr int ATAP  = TILE_CO * ROWP;   // per-tap A tile stride in halves

// ---------------------------------------------------------------------------
// Dequantize int8 (stored int32) weights [COUT][CIN][KW] -> half [KW][COUT][CIN]
// ---------------------------------------------------------------------------
__global__ __launch_bounds__(256) void dequant_kernel(const int* __restrict__ qw,
                                                      const float* __restrict__ scale,
                                                      half_t* __restrict__ wh) {
    const int idx = blockIdx.x * 256 + threadIdx.x;   // flat [k][co][ci]
    const float s = scale[0];
    const int k  = idx >> 18;          // / (512*512)
    const int co = (idx >> 9) & 511;
    const int ci = idx & 511;
    wh[idx] = (half_t)((float)qw[(co * CIN + ci) * KW + k] * s);
}

// ---------------------------------------------------------------------------
// Implicit-GEMM conv1d via V_WMMA_F32_16X16X32_F16, f32 accumulate.
// Grid: (L/128, COUT/128, B). Block: 256 threads = 8 wave32.
// ---------------------------------------------------------------------------
__global__ __launch_bounds__(256) void conv_wmma_kernel(const float*  __restrict__ x,
                                                        const half_t* __restrict__ wh,
                                                        const float*  __restrict__ bias,
                                                        float*        __restrict__ out) {
    __shared__ alignas(16) half_t As[KW * ATAP];      // 3 taps * 128co * 40 halves
    __shared__ alignas(16) half_t Xs[XROWS * ROWP];   // 130 l   * 40 halves, [l][ci]

    const int tid    = threadIdx.x;
    const int lane   = tid & 31;
    const int wav    = tid >> 5;        // 0..7
    const int wy     = wav & 3;         // co direction, 4 waves * 32 rows
    const int wx     = wav >> 2;        // l  direction, 2 waves * 64 cols
    const int lane_m = lane & 15;
    const int hi     = lane >> 4;       // half-wave selector per ISA layouts

    const int lblk   = blockIdx.x;
    const int coblk  = blockIdx.y;
    const int b      = blockIdx.z;
    const int co_base = coblk * TILE_CO;
    const int l_base  = lblk * TILE_L;

    v8f acc[2][4];
    for (int t = 0; t < 2; ++t)
        for (int j = 0; j < 4; ++j)
            acc[t][j] = (v8f){};

    // bias per accumulator element (C/D layout: VGPR r -> M = r + 8*hi)
    float bv[2][8];
    for (int t = 0; t < 2; ++t)
        for (int r = 0; r < 8; ++r)
            bv[t][r] = bias[co_base + wy * 32 + t * 16 + hi * 8 + r];

    const float* xb = x + (size_t)b * CIN * LEN;

    for (int ci0 = 0; ci0 < CIN; ci0 += TILE_CI) {
        __syncthreads();

        // ---- stage weights: 3 taps * 128co * 32ci halves = 1536 16B chunks, 6/thread
        for (int j = 0; j < 6; ++j) {
            const int c = tid + j * 256;
            const int e = c * 8;              // element index in [k][co][ci32]
            const int k  = e >> 12;           // / 4096
            const int r  = e & 4095;
            const int co = r >> 5;
            const int cio = r & 31;
            const v8h* src = reinterpret_cast<const v8h*>(
                wh + ((size_t)(k * COUT + co_base + co) * CIN + ci0 + cio));
            *reinterpret_cast<v8h*>(&As[k * ATAP + co * ROWP + cio]) = *src;
        }

        // ---- stage x slab transposed: Xs[l][ci] = x[b][ci0+ci][l_base-1+l], f32->f16
        for (int i = tid; i < TILE_CI * XROWS; i += 256) {
            const int ci = i / XROWS;
            const int l  = i - ci * XROWS;
            const int gl = l_base - 1 + l;    // PAD = 1
            float v = 0.0f;
            if (gl >= 0 && gl < LEN)
                v = xb[(size_t)(ci0 + ci) * LEN + gl];
            Xs[l * ROWP + ci] = (half_t)v;
        }
        __syncthreads();

        // ---- 3 taps, 8 WMMAs each
        for (int k = 0; k < KW; ++k) {
            union { v16h v; v8h h[2]; } a[2];
            const int ac0 = hi ? 8 : 0;       // ISA A 16x32 f16 layout
            for (int t = 0; t < 2; ++t) {
                const half_t* p = &As[k * ATAP + (wy * 32 + t * 16 + lane_m) * ROWP + ac0];
                a[t].h[0] = *reinterpret_cast<const v8h*>(p);
                a[t].h[1] = *reinterpret_cast<const v8h*>(p + 16);
            }
            const int bc0 = hi ? 8 : 0;       // B 32x16: same K order as A (8h+s, 16+8h+s)
            for (int j = 0; j < 4; ++j) {
                union { v16h v; v8h h[2]; } bf;
                const half_t* p = &Xs[(wx * 64 + j * 16 + lane_m + k) * ROWP + bc0];
                bf.h[0] = *reinterpret_cast<const v8h*>(p);
                bf.h[1] = *reinterpret_cast<const v8h*>(p + 16);
                for (int t = 0; t < 2; ++t) {
                    acc[t][j] = __builtin_amdgcn_wmma_f32_16x16x32_f16(
                        false, a[t].v, false, bf.v, (short)0, acc[t][j], false, false);
                }
            }
        }
    }

    // ---- epilogue: add bias, store fp32
    for (int t = 0; t < 2; ++t) {
        const int co = co_base + wy * 32 + t * 16 + hi * 8;
        for (int j = 0; j < 4; ++j) {
            const int lcol = l_base + wx * 64 + j * 16 + lane_m;
            for (int r = 0; r < 8; ++r) {
                out[((size_t)b * COUT + (co + r)) * LEN + lcol] = acc[t][j][r] + bv[t][r];
            }
        }
    }
}

// ---------------------------------------------------------------------------
extern "C" void kernel_launch(void* const* d_in, const int* in_sizes, int n_in,
                              void* d_out, int out_size, void* d_ws, size_t ws_size,
                              hipStream_t stream) {
    const float* x     = (const float*)d_in[0];
    const int*   qw    = (const int*)d_in[1];
    const float* scale = (const float*)d_in[2];
    const float* bias  = (const float*)d_in[3];
    float*       out   = (float*)d_out;
    half_t*      wh    = (half_t*)d_ws;       // 3*512*512 halves = 1.5 MB scratch

    dequant_kernel<<<dim3((KW * COUT * CIN) / 256), 256, 0, stream>>>(qw, scale, wh);

    dim3 grid(LEN / TILE_L, COUT / TILE_CO, BATCH);
    conv_wmma_kernel<<<grid, 256, 0, stream>>>(x, wh, bias, out);
}
